// DecoderLSTM_new_62783831933634
// MI455X (gfx1250) — matrix operand order repaired and checked by reading, not compiled, not run on hardware
//
#include <hip/hip_runtime.h>
#include <hip/hip_bf16.h>

// ---------------------------------------------------------------------------
// Problem constants
// ---------------------------------------------------------------------------
#define BB   64
#define NN   196
#define HH   512
#define EE   512
#define AA   512
#define VV   20000
#define LL   20
#define STEPS (LL - 1)        // 19
#define KCAT 1536             // E + H + H  ([x | context | h])
#define G4H  2048             // 4*H

// ---------------------------------------------------------------------------
// Types for WMMA (gfx1250, wave32)
// ---------------------------------------------------------------------------
typedef __bf16 bf16_t;
typedef __bf16 v16bf __attribute__((ext_vector_type(16)));
typedef float  v8f   __attribute__((ext_vector_type(8)));
typedef unsigned int u32x4 __attribute__((ext_vector_type(4)));

union ABFrag {
    v16bf v;
    u32x4 u[2];
};

__device__ __forceinline__ unsigned short f2bf(float f) {
    union { float f; unsigned int u; } x;
    x.f = f;
    unsigned int u = x.u;
    unsigned int r = u + 0x7FFFu + ((u >> 16) & 1u);   // round-to-nearest-even
    return (unsigned short)(r >> 16);
}

__device__ __forceinline__ float sigmoidf_(float x) {
    return 1.0f / (1.0f + expf(-x));
}

// ---------------------------------------------------------------------------
// f32 -> bf16 conversion (grid-stride)
// ---------------------------------------------------------------------------
__global__ void k_f32_to_bf16(const float* __restrict__ src,
                              unsigned short* __restrict__ dst, int n) {
    for (int i = blockIdx.x * blockDim.x + threadIdx.x; i < n;
         i += gridDim.x * blockDim.x)
        dst[i] = f2bf(src[i]);
}

// Pack W_ih [2048,1024] and W_hh [2048,512] into Wcat [2048,1536] (bf16)
__global__ void k_build_wcat(const float* __restrict__ W_ih,
                             const float* __restrict__ W_hh,
                             unsigned short* __restrict__ wcat) {
    int n = G4H * KCAT;
    for (int i = blockIdx.x * blockDim.x + threadIdx.x; i < n;
         i += gridDim.x * blockDim.x) {
        int row = i / KCAT;
        int col = i - row * KCAT;
        float v = (col < (EE + HH)) ? W_ih[(size_t)row * (EE + HH) + col]
                                    : W_hh[(size_t)row * HH + (col - EE - HH)];
        wcat[i] = f2bf(v);
    }
}

__global__ void k_zero_f32(float* p, int n) {
    for (int i = blockIdx.x * blockDim.x + threadIdx.x; i < n;
         i += gridDim.x * blockDim.x) p[i] = 0.0f;
}
__global__ void k_zero_u16(unsigned short* p, int n) {
    for (int i = blockIdx.x * blockDim.x + threadIdx.x; i < n;
         i += gridDim.x * blockDim.x) p[i] = 0;
}
// out[:, 0, :] = 0
__global__ void k_zero_out_t0(float* __restrict__ out) {
    int n = BB * VV;
    for (int i = blockIdx.x * blockDim.x + threadIdx.x; i < n;
         i += gridDim.x * blockDim.x) {
        int b = i / VV;
        int v = i - b * VV;
        out[(size_t)b * (LL * VV) + v] = 0.0f;
    }
}

// ---------------------------------------------------------------------------
// Embedding gather for step t: incat[b, 0:512] = bf16(emb[captions[b, t]])
// ---------------------------------------------------------------------------
__global__ void k_embed(const long long* __restrict__ captions,
                        const float* __restrict__ emb,
                        unsigned short* __restrict__ incat, int t) {
    int i = blockIdx.x * blockDim.x + threadIdx.x;
    if (i >= BB * EE) return;
    int b = i >> 9;
    int e = i & (EE - 1);
    long long tok = captions[b * LL + t];
    incat[(size_t)b * KCAT + e] = f2bf(emb[(size_t)tok * EE + e]);
}

// ---------------------------------------------------------------------------
// WMMA GEMM:  C[M,N] (f32) = A[M,K](bf16, row stride lda) @ W[N,K](bf16)^T
//             + bias0[n] + bias1[n]
//
// Branch-free tiling (no guards anywhere -> EXEC stays all-ones around WMMA):
//   block = 128 threads = 4 waves; waves map along M (4 x 16 rows = 64 rows)
//   each wave computes 16(M) x 32(N)  (2 accumulator tiles)
//   grid = (N/32, M/64)   -- exact: N % 32 == 0 and M % 64 == 0 for all calls
//   K unrolled by 64 (K % 64 == 0): 12 b128 loads then 4 WMMAs per iteration
// ---------------------------------------------------------------------------
__global__ void k_wmma_gemm_xwt(const bf16_t* __restrict__ A, int lda,
                                const bf16_t* __restrict__ W,
                                const float* __restrict__ bias0,
                                const float* __restrict__ bias1,
                                float* __restrict__ C, long long ldc,
                                int K) {
    const int lane  = threadIdx.x & 31;
    const int wave  = threadIdx.x >> 5;
    const int l15   = lane & 15;
    const int khalf = lane >> 4;              // 0 for lanes 0-15, 1 for 16-31

    const int m0 = (blockIdx.y * 4 + wave) * 16;
    const int n0 = blockIdx.x * 32;

    // A: lane holds rows m0+l15; 16 contiguous K vals at +16*khalf
    const bf16_t* Arow  = A + (size_t)(m0 + l15) * lda + khalf * 16;
    // B: lane = weight row (output col) n+l15; 16 contiguous K vals at +16*khalf
    const bf16_t* Brow0 = W + (size_t)(n0 + l15) * K + khalf * 16;
    const bf16_t* Brow1 = Brow0 + (size_t)16 * K;

    v8f acc0 = {};
    v8f acc1 = {};

    for (int k0 = 0; k0 < K; k0 += 64) {
        ABFrag a0, a1, b00, b10, b01, b11;
        // issue all loads first; compiler clauses them and overlaps waits
        a0.u[0]  = *(const u32x4*)(Arow + k0);
        a0.u[1]  = *(const u32x4*)(Arow + k0 + 8);
        a1.u[0]  = *(const u32x4*)(Arow + k0 + 32);
        a1.u[1]  = *(const u32x4*)(Arow + k0 + 40);
        b00.u[0] = *(const u32x4*)(Brow0 + k0);
        b00.u[1] = *(const u32x4*)(Brow0 + k0 + 8);
        b10.u[0] = *(const u32x4*)(Brow1 + k0);
        b10.u[1] = *(const u32x4*)(Brow1 + k0 + 8);
        b01.u[0] = *(const u32x4*)(Brow0 + k0 + 32);
        b01.u[1] = *(const u32x4*)(Brow0 + k0 + 40);
        b11.u[0] = *(const u32x4*)(Brow1 + k0 + 32);
        b11.u[1] = *(const u32x4*)(Brow1 + k0 + 40);

        acc0 = __builtin_amdgcn_wmma_f32_16x16x32_bf16(
            false, a0.v, false, b00.v, (short)0, acc0, false, false);
        acc1 = __builtin_amdgcn_wmma_f32_16x16x32_bf16(
            false, a0.v, false, b10.v, (short)0, acc1, false, false);
        acc0 = __builtin_amdgcn_wmma_f32_16x16x32_bf16(
            false, a1.v, false, b01.v, (short)0, acc0, false, false);
        acc1 = __builtin_amdgcn_wmma_f32_16x16x32_bf16(
            false, a1.v, false, b11.v, (short)0, acc1, false, false);
    }

    // store: D element r -> row m0 + r + 8*khalf, col n + l15 (coalesced)
    int n_a = n0 + l15;
    int n_b = n_a + 16;
    float bva = 0.0f, bvb = 0.0f;
    if (bias0) { bva += bias0[n_a]; bvb += bias0[n_b]; }
    if (bias1) { bva += bias1[n_a]; bvb += bias1[n_b]; }
    float* Cbase = C + (size_t)(m0 + khalf * 8) * ldc;
#pragma unroll
    for (int r = 0; r < 8; ++r) {
        float* crow = Cbase + (size_t)r * ldc;
        crow[n_a] = acc0[r] + bva;
        crow[n_b] = acc1[r] + bvb;
    }
}

// ---------------------------------------------------------------------------
// Attention scores: scores[b,n] = sum_a tanh(enc_proj[b,n,a] + dec[b,a])*eW[a] + eb
// grid (NN, BB), block 128
// ---------------------------------------------------------------------------
__global__ void k_attn_scores(const float* __restrict__ enc_proj,
                              const float* __restrict__ dec,
                              const float* __restrict__ eW,
                              const float* __restrict__ eb,
                              float* __restrict__ scores) {
    __shared__ float red[128];
    int n = blockIdx.x, b = blockIdx.y, tid = threadIdx.x;
    const float* ep = enc_proj + ((size_t)b * NN + n) * AA;
    const float* dv = dec + (size_t)b * AA;
    float s = 0.0f;
    for (int a = tid; a < AA; a += 128)
        s += tanhf(ep[a] + dv[a]) * eW[a];
    red[tid] = s;
    __syncthreads();
    for (int off = 64; off > 0; off >>= 1) {
        if (tid < off) red[tid] += red[tid + off];
        __syncthreads();
    }
    if (tid == 0) scores[b * NN + n] = red[0] + eb[0];
}

// ---------------------------------------------------------------------------
// Softmax over N + context = attw @ encoder_outputs, written bf16 into
// incat[b, 512:1024].  grid BB, block 256.
// ---------------------------------------------------------------------------
__global__ void k_softmax_ctx(const float* __restrict__ scores,
                              const float* __restrict__ enc,
                              unsigned short* __restrict__ incat) {
    __shared__ float sm[256];
    __shared__ float attw[NN];
    int b = blockIdx.x, tid = threadIdx.x;

    float v = (tid < NN) ? scores[b * NN + tid] : -3.4e38f;
    sm[tid] = v;
    __syncthreads();
    for (int off = 128; off > 0; off >>= 1) {
        if (tid < off) sm[tid] = fmaxf(sm[tid], sm[tid + off]);
        __syncthreads();
    }
    float mx = sm[0];
    __syncthreads();

    float e = (tid < NN) ? expf(v - mx) : 0.0f;
    sm[tid] = e;
    __syncthreads();
    for (int off = 128; off > 0; off >>= 1) {
        if (tid < off) sm[tid] += sm[tid + off];
        __syncthreads();
    }
    float inv = 1.0f / sm[0];
    __syncthreads();
    if (tid < NN) attw[tid] = e * inv;
    __syncthreads();

    for (int h = tid; h < HH; h += 256) {
        float c = 0.0f;
        const float* ebase = enc + ((size_t)b * NN) * HH + h;
        for (int n = 0; n < NN; ++n)
            c += attw[n] * ebase[(size_t)n * HH];
        incat[(size_t)b * KCAT + EE + h] = f2bf(c);
    }
}

// ---------------------------------------------------------------------------
// LSTM cell: gates [B,2048] (i|f|g|o), c state f32; h -> bf16 incat[b,1024:1536]
// ---------------------------------------------------------------------------
__global__ void k_lstm_cell(const float* __restrict__ gates,
                            float* __restrict__ c,
                            unsigned short* __restrict__ incat) {
    int i = blockIdx.x * blockDim.x + threadIdx.x;
    if (i >= BB * HH) return;
    int b = i >> 9;
    int u = i & (HH - 1);
    const float* g = gates + (size_t)b * G4H;
    float gi = g[u];
    float gf = g[HH + u];
    float gg = g[2 * HH + u];
    float go = g[3 * HH + u];
    float cn = sigmoidf_(gf) * c[i] + sigmoidf_(gi) * tanhf(gg);
    float hn = sigmoidf_(go) * tanhf(cn);
    c[i] = cn;
    incat[(size_t)b * KCAT + 2 * HH + u] = f2bf(hn);
}

// ---------------------------------------------------------------------------
// Host-side orchestration
// ---------------------------------------------------------------------------
static inline dim3 gemm_grid(int M, int N) {
    // exact cover: N % 32 == 0, M % 64 == 0 for every call below
    return dim3((unsigned)(N / 32), (unsigned)(M / 64), 1);
}

extern "C" void kernel_launch(void* const* d_in, const int* in_sizes, int n_in,
                              void* d_out, int out_size, void* d_ws, size_t ws_size,
                              hipStream_t stream) {
    const float*     enc_out  = (const float*)d_in[0];
    const long long* captions = (const long long*)d_in[1];
    const float*     emb      = (const float*)d_in[2];
    const float*     W_ih     = (const float*)d_in[3];
    const float*     W_hh     = (const float*)d_in[4];
    const float*     b_ih     = (const float*)d_in[5];
    const float*     b_hh     = (const float*)d_in[6];
    const float*     enc_W    = (const float*)d_in[7];
    const float*     enc_b    = (const float*)d_in[8];
    const float*     dec_W    = (const float*)d_in[9];
    const float*     dec_b    = (const float*)d_in[10];
    const float*     energy_W = (const float*)d_in[11];
    const float*     energy_b = (const float*)d_in[12];
    const float*     fc_W     = (const float*)d_in[13];
    const float*     fc_b     = (const float*)d_in[14];
    float* out = (float*)d_out;

    // ---- carve workspace -------------------------------------------------
    char* p = (char*)d_ws;
    auto carve = [&](size_t bytes) -> char* {
        char* r = p;
        p += (bytes + 255) & ~(size_t)255;
        return r;
    };
    unsigned short* enc_bf   = (unsigned short*)carve((size_t)BB * NN * HH * 2);
    unsigned short* encW_bf  = (unsigned short*)carve((size_t)AA * HH * 2);
    unsigned short* decW_bf  = (unsigned short*)carve((size_t)AA * HH * 2);
    unsigned short* fcW_bf   = (unsigned short*)carve((size_t)VV * HH * 2);
    unsigned short* wcat_bf  = (unsigned short*)carve((size_t)G4H * KCAT * 2);
    float*          enc_proj = (float*)carve((size_t)BB * NN * AA * 4);
    float*          dec      = (float*)carve((size_t)BB * AA * 4);
    float*          scores   = (float*)carve((size_t)BB * NN * 4);
    unsigned short* incat    = (unsigned short*)carve((size_t)BB * KCAT * 2);
    float*          gates    = (float*)carve((size_t)BB * G4H * 4);
    float*          cstate   = (float*)carve((size_t)BB * HH * 4);

    // ---- one-time conversions / packing ---------------------------------
    {
        int n;
        n = BB * NN * HH;
        k_f32_to_bf16<<<dim3((n + 255) / 256), dim3(256), 0, stream>>>(enc_out, enc_bf, n);
        n = AA * HH;
        k_f32_to_bf16<<<dim3((n + 255) / 256), dim3(256), 0, stream>>>(enc_W, encW_bf, n);
        k_f32_to_bf16<<<dim3((n + 255) / 256), dim3(256), 0, stream>>>(dec_W, decW_bf, n);
        n = VV * HH;
        k_f32_to_bf16<<<dim3((n + 255) / 256), dim3(256), 0, stream>>>(fc_W, fcW_bf, n);
        n = G4H * KCAT;
        k_build_wcat<<<dim3((n + 255) / 256), dim3(256), 0, stream>>>(W_ih, W_hh, wcat_bf);

        k_zero_f32<<<dim3((BB * HH + 255) / 256), dim3(256), 0, stream>>>(cstate, BB * HH);
        k_zero_u16<<<dim3((BB * KCAT + 255) / 256), dim3(256), 0, stream>>>(incat, BB * KCAT);
        k_zero_out_t0<<<dim3((BB * VV + 255) / 256), dim3(256), 0, stream>>>(out);
    }

    // ---- enc_proj = encoder_outputs @ enc_W^T + enc_b  (12544x512x512) ---
    k_wmma_gemm_xwt<<<gemm_grid(BB * NN, AA), dim3(128), 0, stream>>>(
        (const bf16_t*)enc_bf, HH, (const bf16_t*)encW_bf,
        enc_b, nullptr, enc_proj, AA, HH);

    // ---- scan over 19 steps ---------------------------------------------
    const bf16_t* h_bf = (const bf16_t*)(incat) + 2 * HH;  // h slice, lda=KCAT

    for (int t = 0; t < STEPS; ++t) {
        // x_t gather into incat[:, 0:512]
        k_embed<<<dim3((BB * EE + 255) / 256), dim3(256), 0, stream>>>(
            captions, emb, incat, t);

        // dec = h @ dec_W^T + dec_b   (64x512x512)
        k_wmma_gemm_xwt<<<gemm_grid(BB, AA), dim3(128), 0, stream>>>(
            h_bf, KCAT, (const bf16_t*)decW_bf,
            dec_b, nullptr, dec, AA, HH);

        // attention scores + softmax + context
        k_attn_scores<<<dim3(NN, BB), dim3(128), 0, stream>>>(
            enc_proj, dec, energy_W, energy_b, scores);
        k_softmax_ctx<<<dim3(BB), dim3(256), 0, stream>>>(
            scores, enc_out, incat);

        // gates = [x|ctx|h] @ Wcat^T + b_ih + b_hh   (64x2048x1536)
        k_wmma_gemm_xwt<<<gemm_grid(BB, G4H), dim3(128), 0, stream>>>(
            (const bf16_t*)incat, KCAT, (const bf16_t*)wcat_bf,
            b_ih, b_hh, gates, G4H, KCAT);

        // LSTM cell -> updates c, writes h (bf16) into incat[:, 1024:1536]
        k_lstm_cell<<<dim3((BB * HH + 255) / 256), dim3(256), 0, stream>>>(
            gates, cstate, incat);

        // logits = h @ fc_W^T + fc_b   (64x20000x512) -> out[:, t+1, :]
        k_wmma_gemm_xwt<<<gemm_grid(BB, VV), dim3(128), 0, stream>>>(
            h_bf, KCAT, (const bf16_t*)fcW_bf,
            fc_b, nullptr, out + (size_t)(t + 1) * VV, (long long)LL * VV,
            HH);
    }
}
